// MogLSTM_28827820491443
// MI455X (gfx1250) — compile-verified, matrix operand order checked by reading, not run on hardware
//
#include <hip/hip_runtime.h>
#include <hip/hip_bf16.h>

// Bidirectional LSTM for MI455X (gfx1250, wave32, WMMA).
// Phase 1: bf16 conversion of x and weights, bias pre-sum.
// Phase 2: WMMA GEMM  G = x @ Wih.T + (bih+bhh), 64x64 tile per wave,
//          output stored in *fragment-native* order so phase 3 reads it with
//          vector loads straight into WMMA accumulators.
// Phase 3: sequential recurrence, 1 workgroup (32 waves) per direction,
//          h kept in LDS as a zero-padded 16x512 bf16 WMMA A-tile,
//          each wave owns a 16-wide j-slice and all 4 gate quadrants so the
//          LSTM cell update is register-resident (c never leaves VGPRs).
//
// Fragment-native G layout: a 16-row M tile (m = t*8+b) holds t=2*m16 in rows
// 0..7 and t=2*m16+1 in rows 8..15, exactly the WMMA C/D register layout.
//   G[( (t>>1)*2048 + n )*16 + (t&1)*8 + b]

typedef __bf16 bf16_t;
typedef __attribute__((ext_vector_type(16))) __bf16 v16bf;
typedef __attribute__((ext_vector_type(8)))  __bf16 v8bf;
typedef __attribute__((ext_vector_type(8)))  float  v8f;

#define T_LEN 2048
#define B_SZ  8
#define I_SZ  512
#define H_SZ  512
#define G4H   2048   // 4*H

// ---------------- prep kernels ----------------

__global__ void k_f32_to_bf16(const float* __restrict__ src,
                              bf16_t* __restrict__ dst, int n) {
  int i = blockIdx.x * blockDim.x + threadIdx.x;
  int stride = gridDim.x * blockDim.x;
  for (; i < n; i += stride) dst[i] = (bf16_t)src[i];
}

__global__ void k_bias_sum(const float* __restrict__ a1, const float* __restrict__ a2,
                           float* __restrict__ o1,
                           const float* __restrict__ b1, const float* __restrict__ b2,
                           float* __restrict__ o2, int n) {
  int i = blockIdx.x * blockDim.x + threadIdx.x;
  if (i < n) { o1[i] = a1[i] + a2[i]; o2[i] = b1[i] + b2[i]; }
}

// ---------------- phase 2: gate pre-activation GEMM ----------------
// 64x64 tile per wave: 4 M-subtiles x 4 N-subtiles = 16 accumulators.
// Each B fragment feeds 4 WMMAs, each A fragment feeds 4 WMMAs
// (16 WMMAs : 16 b128 loads per k-step of 32).
__global__ void __launch_bounds__(256)
k_gate_gemm(const bf16_t* __restrict__ X,
            const bf16_t* __restrict__ Wf, const bf16_t* __restrict__ Wb,
            const float* __restrict__ biasf, const float* __restrict__ biasb,
            float* __restrict__ Gf, float* __restrict__ Gb) {
  const int lane = threadIdx.x & 31;
  const int wave = blockIdx.x * (blockDim.x >> 5) + (threadIdx.x >> 5);
  const int dir  = wave >> 13;          // 8192 waves per direction
  const int rem  = wave & 8191;
  const int mt64 = rem >> 5;            // 0..255  (M tile of 64)
  const int n0   = (rem & 31) * 64;     // 0..2047 (N tile of 64)
  const int lm   = lane & 15;
  const int hi   = lane >> 4;
  const int koff = hi * 8;

  const bf16_t* W    = dir ? Wb : Wf;
  const float*  bias = dir ? biasb : biasf;
  float*        G    = dir ? Gb : Gf;

  const bf16_t* arow[4];
#pragma unroll
  for (int s = 0; s < 4; ++s) {
    const int m = (mt64 * 4 + s) * 16 + lm;   // row index = t*8 + b
    const int t = m >> 3, b = m & 7;
    arow[s] = X + ((size_t)(b * T_LEN + t)) * I_SZ;
  }

  const bf16_t* brow[4];
  v8f acc[4][4];
#pragma unroll
  for (int q = 0; q < 4; ++q) {
    const int n = n0 + q * 16 + lm;
    brow[q] = W + (size_t)n * I_SZ;
    const float bv = bias[n];
#pragma unroll
    for (int s = 0; s < 4; ++s)
#pragma unroll
      for (int r = 0; r < 8; ++r) acc[s][q][r] = bv;  // bias broadcast down M
  }

  for (int k0 = 0; k0 < I_SZ; k0 += 32) {
    union { v16bf v; v8bf h[2]; } A[4];
#pragma unroll
    for (int s = 0; s < 4; ++s) {
      A[s].h[0] = *(const v8bf*)(arow[s] + k0 + koff);
      A[s].h[1] = *(const v8bf*)(arow[s] + k0 + 16 + koff);
    }
#pragma unroll
    for (int q = 0; q < 4; ++q) {
      union { v16bf v; v8bf h[2]; } Bm;
      Bm.h[0] = *(const v8bf*)(brow[q] + k0 + koff);
      Bm.h[1] = *(const v8bf*)(brow[q] + k0 + 16 + koff);
#pragma unroll
      for (int s = 0; s < 4; ++s)
        acc[s][q] = __builtin_amdgcn_wmma_f32_16x16x32_bf16(
            false, A[s].v, false, Bm.v, (short)0, acc[s][q], false, false);
    }
  }

  // fragment-native store: 8 consecutive floats (r=0..7) per lane per acc
#pragma unroll
  for (int s = 0; s < 4; ++s)
#pragma unroll
    for (int q = 0; q < 4; ++q) {
      const size_t base =
          ((size_t)(mt64 * 4 + s) * G4H + (n0 + q * 16 + lm)) * 16 + hi * 8;
      *(v8f*)(G + base) = acc[s][q];
    }
}

// ---------------- phase 3: sequential recurrence ----------------
// grid = 2 blocks (dir), 1024 threads = 32 waves.  Wave w owns hidden columns
// j0 = 16*w and computes gate quadrants i,f,g,o for those columns -> the cell
// update is purely register-local.  h lives in LDS as a 16x512 bf16 A-tile
// (rows 8..15 permanently zero).  c lives in VGPRs for the whole time loop.
__global__ void __launch_bounds__(1024, 1)
k_lstm_seq(const float* __restrict__ Gf, const float* __restrict__ Gb,
           const bf16_t* __restrict__ Whhf, const bf16_t* __restrict__ Whhb,
           float* __restrict__ out) {
  __shared__ bf16_t h_lds[16][H_SZ];   // 16 KB

  const int dir  = blockIdx.x;
  const float*  G = dir ? Gb : Gf;
  const bf16_t* W = dir ? Whhb : Whhf;

  const int lane = threadIdx.x & 31;
  const int wv   = threadIdx.x >> 5;    // 0..31
  const int j0   = wv * 16;
  const int lm   = lane & 15;
  const int hi   = lane >> 4;
  const int koff = hi * 8;

  for (int i = threadIdx.x; i < 16 * H_SZ; i += blockDim.x)
    (&h_lds[0][0])[i] = (bf16_t)0.0f;

  float c[8];
#pragma unroll
  for (int r = 0; r < 8; ++r) c[r] = 0.0f;

  const bf16_t* brow[4];
#pragma unroll
  for (int q = 0; q < 4; ++q)
    brow[q] = W + (size_t)(q * H_SZ + j0 + lm) * H_SZ;
  const bf16_t* arow = &h_lds[lm][0];

  __syncthreads();

  for (int step = 0; step < T_LEN; ++step) {
    const int t = dir ? (T_LEN - 1 - step) : step;

    // init accumulators from fragment-native G (x-part + biases): one v8f
    // vector load per gate quadrant instead of 8 scalar gathers
    v8f acc[4];
#pragma unroll
    for (int q = 0; q < 4; ++q) {
      if (hi == 0) {
        acc[q] = *(const v8f*)(G + ((size_t)(t >> 1) * G4H + q * H_SZ + j0 + lm) * 16
                                   + (t & 1) * 8);
      } else {
#pragma unroll
        for (int r = 0; r < 8; ++r) acc[q][r] = 0.0f;   // padding rows M=8..15
      }
    }

    // prefetch next step's gate fragments into cache while we do the WMMAs
    const int tn = dir ? (t - 1) : (t + 1);
    if (tn >= 0 && tn < T_LEN && hi == 0)
      __builtin_prefetch(G + ((size_t)(tn >> 1) * G4H + j0 + lm) * 16 + (tn & 1) * 8,
                         0, 1);

#pragma unroll 4
    for (int k0 = 0; k0 < H_SZ; k0 += 32) {
      union { v16bf v; v8bf h[2]; } A;
      A.h[0] = *(const v8bf*)(arow + k0 + koff);
      A.h[1] = *(const v8bf*)(arow + k0 + 16 + koff);
#pragma unroll
      for (int q = 0; q < 4; ++q) {
        union { v16bf v; v8bf h[2]; } Bm;
        Bm.h[0] = *(const v8bf*)(brow[q] + k0 + koff);
        Bm.h[1] = *(const v8bf*)(brow[q] + k0 + 16 + koff);
        acc[q] = __builtin_amdgcn_wmma_f32_16x16x32_bf16(
            false, A.v, false, Bm.v, (short)0, acc[q], false, false);
      }
    }

    __syncthreads();   // everyone done reading h_lds for this step

#pragma unroll
    for (int r = 0; r < 8; ++r) {
      const float ig = 1.0f / (1.0f + __expf(-acc[0][r]));
      const float fg = 1.0f / (1.0f + __expf(-acc[1][r]));
      const float gg = tanhf(acc[2][r]);
      const float og = 1.0f / (1.0f + __expf(-acc[3][r]));
      const float cn = fg * c[r] + ig * gg;
      c[r] = cn;
      const float hn = og * tanhf(cn);
      if (hi == 0) {                                   // valid rows b = r (0..7)
        h_lds[r][j0 + lm] = (bf16_t)hn;
        out[(size_t)(t * B_SZ + r) * (2 * H_SZ) + dir * H_SZ + j0 + lm] = hn;
      }
    }

    __syncthreads();   // h_lds fully updated before next step's reads
  }
}

// ---------------- host launcher ----------------

extern "C" void kernel_launch(void* const* d_in, const int* in_sizes, int n_in,
                              void* d_out, int out_size, void* d_ws, size_t ws_size,
                              hipStream_t stream) {
  const float* input = (const float*)d_in[0];
  const float* Wih_f = (const float*)d_in[1];
  const float* Whh_f = (const float*)d_in[2];
  const float* bih_f = (const float*)d_in[3];
  const float* bhh_f = (const float*)d_in[4];
  const float* Wih_b = (const float*)d_in[5];
  const float* Whh_b = (const float*)d_in[6];
  const float* bih_b = (const float*)d_in[7];
  const float* bhh_b = (const float*)d_in[8];
  float* out = (float*)d_out;

  char* ws = (char*)d_ws;
  size_t off = 0;
  auto carve = [&](size_t bytes) -> char* {
    char* p = ws + off;
    off = (off + bytes + 255) & ~(size_t)255;
    return p;
  };
  float*  Gf    = (float*)carve((size_t)T_LEN * B_SZ * G4H * sizeof(float)); // 128 MB
  float*  Gb    = (float*)carve((size_t)T_LEN * B_SZ * G4H * sizeof(float)); // 128 MB
  bf16_t* Xbf   = (bf16_t*)carve((size_t)B_SZ * T_LEN * I_SZ * 2);           // 16 MB
  bf16_t* Wihf  = (bf16_t*)carve((size_t)G4H * I_SZ * 2);
  bf16_t* Whhf2 = (bf16_t*)carve((size_t)G4H * H_SZ * 2);
  bf16_t* Wihb  = (bf16_t*)carve((size_t)G4H * I_SZ * 2);
  bf16_t* Whhb2 = (bf16_t*)carve((size_t)G4H * H_SZ * 2);
  float*  biasf = (float*)carve((size_t)G4H * sizeof(float));
  float*  biasb = (float*)carve((size_t)G4H * sizeof(float));

  // phase 1: conversions + bias pre-sum
  k_f32_to_bf16<<<4096, 256, 0, stream>>>(input, Xbf, B_SZ * T_LEN * I_SZ);
  k_f32_to_bf16<<<1024, 256, 0, stream>>>(Wih_f, Wihf,  G4H * I_SZ);
  k_f32_to_bf16<<<1024, 256, 0, stream>>>(Whh_f, Whhf2, G4H * H_SZ);
  k_f32_to_bf16<<<1024, 256, 0, stream>>>(Wih_b, Wihb,  G4H * I_SZ);
  k_f32_to_bf16<<<1024, 256, 0, stream>>>(Whh_b, Whhb2, G4H * H_SZ);
  k_bias_sum<<<(G4H + 255) / 256, 256, 0, stream>>>(bih_f, bhh_f, biasf,
                                                    bih_b, bhh_b, biasb, G4H);

  // phase 2: 2 dirs x (16384/64 M-tiles) x (2048/64 N-tiles) = 16384 waves
  k_gate_gemm<<<2048, 256, 0, stream>>>(Xbf, Wihf, Wihb, biasf, biasb, Gf, Gb);

  // phase 3: one workgroup per direction
  k_lstm_seq<<<2, 1024, 0, stream>>>(Gf, Gb, Whhf2, Whhb2, out);
}
